// AGCR_59983513255964
// MI455X (gfx1250) — matrix operand order repaired and checked by reading, not compiled, not run on hardware
//
#include <hip/hip_runtime.h>
#include <stdint.h>

typedef __attribute__((ext_vector_type(16))) __bf16          v16bf;
typedef __attribute__((ext_vector_type(16))) unsigned short  v16u;
typedef __attribute__((ext_vector_type(8)))  float           v8f;

typedef __attribute__((__vector_size__(4 * sizeof(int)))) int i32x4;
typedef __attribute__((address_space(1))) i32x4 g_i32x4;   // global int4
typedef __attribute__((address_space(3))) i32x4 l_i32x4;   // LDS int4

#define B_   8
#define C_   512
#define N_   4096
#define CQK  128
#define TOPK 409   // int(4096 * 0.1)

__device__ __forceinline__ unsigned short f2bf(float x) {
    unsigned u = __float_as_uint(x);
    unsigned r = ((u >> 16) & 1u) + 0x7FFFu;   // round-to-nearest-even
    return (unsigned short)((u + r) >> 16);
}
__device__ __forceinline__ unsigned pack2(float lo, float hi) {
    return (unsigned)f2bf(lo) | ((unsigned)f2bf(hi) << 16);
}

__device__ __forceinline__ float wredsum(float v) {
#pragma unroll
    for (int m = 16; m; m >>= 1) v += __shfl_xor(v, m, 32);
    return v;
}
__device__ __forceinline__ int wredsumi(int v) {
#pragma unroll
    for (int m = 16; m; m >>= 1) v += __shfl_xor(v, m, 32);
    return v;
}
__device__ __forceinline__ float wredmax(float v) {
#pragma unroll
    for (int m = 16; m; m >>= 1) v = fmaxf(v, __shfl_xor(v, m, 32));
    return v;
}

// order-preserving float <-> uint key (ascending)
__device__ __forceinline__ unsigned okey(float s) {
    unsigned u = __float_as_uint(s);
    return (u & 0x80000000u) ? ~u : (u ^ 0x80000000u);
}
__device__ __forceinline__ float okeyinv(unsigned k) {
    unsigned u = (k & 0x80000000u) ? (k ^ 0x80000000u) : ~k;
    return __uint_as_float(u);
}

// ---------------- conversion kernels (vectorized: 8 elems / thread-iter) ----------------
__global__ void cvt_bf16_v8(const float* __restrict__ in, unsigned short* __restrict__ out,
                            int n8) {
    int i = blockIdx.x * blockDim.x + threadIdx.x;
    int stride = gridDim.x * blockDim.x;
    for (; i < n8; i += stride) {
        size_t base = (size_t)i * 8;
        float4 a = *(const float4*)(in + base);
        float4 b = *(const float4*)(in + base + 4);
        uint4 o;
        o.x = pack2(a.x, a.y); o.y = pack2(a.z, a.w);
        o.z = pack2(b.x, b.y); o.w = pack2(b.z, b.w);
        *(uint4*)(out + base) = o;
    }
}
// Wf is (512 x 1024) row-major; take first 512 columns
__global__ void cvt_wf1_v8(const float* __restrict__ Wf, unsigned short* __restrict__ out) {
    int i = blockIdx.x * blockDim.x + threadIdx.x;
    int stride = gridDim.x * blockDim.x;
    for (; i < C_ * 64; i += stride) {          // 512 rows * 64 chunks of 8
        int o = i >> 6, seg = i & 63;
        const float* src = Wf + (size_t)o * 1024 + seg * 8;
        float4 a = *(const float4*)(src);
        float4 b = *(const float4*)(src + 4);
        uint4 v;
        v.x = pack2(a.x, a.y); v.y = pack2(a.z, a.w);
        v.z = pack2(b.x, b.y); v.w = pack2(b.z, b.w);
        *(uint4*)(out + (size_t)o * C_ + seg * 8) = v;
    }
}
__global__ void zero_f32(float* __restrict__ p, int n) {
    int i = blockIdx.x * blockDim.x + threadIdx.x;
    int stride = gridDim.x * blockDim.x;
    for (; i < n; i += stride) p[i] = 0.0f;
}

// ---------------- WMMA GEMM:  out[b,o,n] = sum_c W[o,c] * f[b,c,n] ----------------
// block = 256 threads (8 waves); block tile = 16(M) x 512(N); wave tile = 16 x 64
template <bool BF16OUT>
__global__ __launch_bounds__(256)
void gemm_wmma(const unsigned short* __restrict__ Wbf,   // O x 512 (row-major)
               const unsigned short* __restrict__ fbf,   // B x 512 x N (n contiguous)
               void* __restrict__ outp, int O) {
    __shared__ unsigned short lA[16 * 40];    // [m][k], rows 16B-aligned
    __shared__ unsigned short lB[512 * 36];   // [n][k], transposed stage

    const int tid = threadIdx.x, lane = tid & 31, wave = tid >> 5;
    const int ntiles = N_ / 512, otiles = O / 16;
    int blk = blockIdx.x;
    int b  = blk / (otiles * ntiles);
    int r2 = blk % (otiles * ntiles);
    int o0 = (r2 / ntiles) * 16;
    int n0 = (r2 % ntiles) * 512;
    const unsigned short* fb = fbf + (size_t)b * C_ * N_;

    const int lm = lane & 15;
    const int ah = (lane < 16) ? 0 : 8;     // A K-half select per lane group
    const int bh = (lane < 16) ? 0 : 16;    // unused; B reads use ah, same k order as A
    const int col0 = wave * 64 + lm;

    v8f acc[4] = {v8f{}, v8f{}, v8f{}, v8f{}};

    for (int ks = 0; ks < C_; ks += 32) {
        // A tile: 16 x 32 = 64 contiguous b128 chunks -> async global->LDS DMA
        if (tid < 64) {
            int m = tid >> 2, seg = tid & 3;
            const unsigned short* gp = &Wbf[(size_t)(o0 + m) * C_ + ks + seg * 8];
            unsigned short* lp = &lA[m * 40 + seg * 8];
#if __has_builtin(__builtin_amdgcn_global_load_async_to_lds_b128)
            __builtin_amdgcn_global_load_async_to_lds_b128(
                (g_i32x4*)(uintptr_t)gp,
                (l_i32x4*)(unsigned)(uintptr_t)lp, 0, 0);
#else
            *(uint4*)lp = *(const uint4*)gp;
#endif
        }
        // B tile: 32(k) x 512(n); vector global loads, transposed LDS scatter
        for (int it = 0; it < 8; ++it) {
            int id = tid + it * 256;
            int kk = id >> 6, n8 = (id & 63) * 8;
            uint4 v = *(const uint4*)&fb[(size_t)(ks + kk) * N_ + n0 + n8];
            const unsigned short* e16 = (const unsigned short*)&v;
#pragma unroll
            for (int e = 0; e < 8; ++e) lB[(n8 + e) * 36 + kk] = e16[e];
        }
        if (ks + 32 < C_)
            __builtin_prefetch(fb + (size_t)(ks + 32) * N_ + n0 + tid * 16, 0, 1);
#if __has_builtin(__builtin_amdgcn_global_load_async_to_lds_b128)
#if __has_builtin(__builtin_amdgcn_s_wait_asynccnt)
        __builtin_amdgcn_s_wait_asynccnt(0);
#else
        asm volatile("s_wait_asynccnt 0x0" ::: "memory");
#endif
#endif
        __syncthreads();

        v16u au;
#pragma unroll
        for (int e = 0; e < 8; ++e) {
            au[e]     = lA[lm * 40 + ah + e];
            au[8 + e] = lA[lm * 40 + 16 + ah + e];
        }
        v16bf a = __builtin_bit_cast(v16bf, au);
#pragma unroll
        for (int t = 0; t < 4; ++t) {
            v16u bu;
#pragma unroll
            for (int e = 0; e < 16; ++e)
                bu[e] = lB[(col0 + t * 16) * 36 + ((e & 8) << 1) + ah + (e & 7)];
            acc[t] = __builtin_amdgcn_wmma_f32_16x16x32_bf16(false, a, false,
                       __builtin_bit_cast(v16bf, bu), (short)0, acc[t], false, false);
        }
        __syncthreads();
    }

    const int mrow = (lane < 16) ? 0 : 8;
    const int ncol = lane & 15;
#pragma unroll
    for (int r = 0; r < 8; ++r) {
        size_t base = ((size_t)b * O + (o0 + mrow + r)) * N_ + n0 + wave * 64 + ncol;
#pragma unroll
        for (int t = 0; t < 4; ++t) {
            if (BF16OUT) ((unsigned short*)outp)[base + t * 16] = f2bf(acc[t][r]);
            else         ((float*)outp)[base + t * 16]          = acc[t][r];
        }
    }
}

// ---------------- streaming attention: scores + softmax + top-k mean + column sums -------
// One block owns a 16-row strip of attn for one batch; the 16x4096 score strip lives in LDS.
__global__ __launch_bounds__(256)
void attn_kernel(const unsigned short* __restrict__ qbf,   // B x 128 x N (bf16)
                 const unsigned short* __restrict__ kbf,   // B x 128 x N (bf16)
                 float* __restrict__ colsum,               // B x N (pre-zeroed)
                 float* __restrict__ softtopk) {           // B x N
    extern __shared__ __align__(16) char smem[];
    unsigned short* kT = (unsigned short*)smem;                    // [128][132] transposed [j][o]
    unsigned short* qT = kT + 128 * 132;                           // [16][132]  transposed [i][o]
    float* S       = (float*)(smem + (128 * 132 + 16 * 132) * 2);  // [16][4100]
    float* rowmax  = S + 16 * 4100;
    float* rowinvZ = rowmax + 16;

    const int tid = threadIdx.x, lane = tid & 31, wave = tid >> 5;
    const int b  = blockIdx.x >> 8;        // N_/16 = 256 strips per batch
    const int i0 = (blockIdx.x & 255) * 16;
    const unsigned short* qb = qbf + (size_t)b * CQK * N_;
    const unsigned short* kb = kbf + (size_t)b * CQK * N_;

    // stage q strip transposed: qT[i][o]  (one b128 load + 8-way scatter per thread)
    {
        int o = tid >> 1, h = tid & 1;
        uint4 v = *(const uint4*)&qb[(size_t)o * N_ + i0 + h * 8];
        const unsigned short* e16 = (const unsigned short*)&v;
#pragma unroll
        for (int e = 0; e < 8; ++e) qT[(h * 8 + e) * 132 + o] = e16[e];
    }
    __syncthreads();

    const int lm = lane & 15;
    const int ah = (lane < 16) ? 0 : 8;
    const int bh = (lane < 16) ? 0 : 16;

    // A fragments for full K=128 (4 x 16x16x32), loaded once
    v16u au[4];
#pragma unroll
    for (int kk = 0; kk < 4; ++kk)
#pragma unroll
        for (int e = 0; e < 8; ++e) {
            au[kk][e]     = qT[lm * 132 + kk * 32 + ah + e];
            au[kk][8 + e] = qT[lm * 132 + kk * 32 + 16 + ah + e];
        }

    const float sc = 0.088388347648318447f;   // 1/sqrt(128)

    for (int jc = 0; jc < N_; jc += 128) {
        // stage k chunk transposed: kT[j][o] (b128 global load + 8-way LDS scatter)
#pragma unroll
        for (int it = 0; it < 8; ++it) {
            int id = tid + it * 256;
            int o = id >> 4, j8 = (id & 15) * 8;
            uint4 v = *(const uint4*)&kb[(size_t)o * N_ + jc + j8];
            const unsigned short* e16 = (const unsigned short*)&v;
#pragma unroll
            for (int e = 0; e < 8; ++e) kT[(j8 + e) * 132 + o] = e16[e];
        }
        __syncthreads();

        const int jl = wave * 16;
        v8f acc = {};
#pragma unroll
        for (int kk = 0; kk < 4; ++kk) {
            v16u bu;
#pragma unroll
            for (int e = 0; e < 16; ++e)
                bu[e] = kT[(jl + lm) * 132 + kk * 32 + ((e & 8) << 1) + ah + (e & 7)];   // table order, matches A
            acc = __builtin_amdgcn_wmma_f32_16x16x32_bf16(false,
                    __builtin_bit_cast(v16bf, au[kk]), false,
                    __builtin_bit_cast(v16bf, bu), (short)0, acc, false, false);
        }
        const int mrow = (lane < 16) ? 0 : 8;
#pragma unroll
        for (int r = 0; r < 8; ++r)
            S[(mrow + r) * 4100 + jc + jl + lm] = acc[r] * sc;
        __syncthreads();
    }

    // per-row softmax stats + top-409 mean (each wave owns 2 rows)
    for (int rr = wave * 2; rr < wave * 2 + 2; ++rr) {
        const float* Sr = S + rr * 4100;
        float mx = -3.4e38f;
        for (int j = lane; j < N_; j += 32) mx = fmaxf(mx, Sr[j]);
        mx = wredmax(mx);
        float Z = 0.f;
        for (int j = lane; j < N_; j += 32) Z += __expf(Sr[j] - mx);
        Z = wredsum(Z);

        // bisect largest key T with count(key >= T) >= TOPK
        unsigned lo = 0u, hi = 0xFFFFFFFFu;
        while (lo < hi) {
            unsigned mid = lo + ((hi - lo) >> 1) + 1u;
            int c = 0;
            for (int j = lane; j < N_; j += 32) c += (okey(Sr[j]) >= mid) ? 1 : 0;
            c = wredsumi(c);
            if (c >= TOPK) lo = mid; else hi = mid - 1u;
        }
        int cg = 0; float sg = 0.f;
        for (int j = lane; j < N_; j += 32) {
            float s = Sr[j];
            if (okey(s) > lo) { cg++; sg += __expf(s - mx); }
        }
        cg = wredsumi(cg); sg = wredsum(sg);
        float stk = (sg + (float)(TOPK - cg) * __expf(okeyinv(lo) - mx))
                    / ((float)TOPK * Z);
        if (lane == 0) {
            softtopk[(size_t)b * N_ + i0 + rr] = stk;
            rowmax[rr]  = mx;
            rowinvZ[rr] = 1.0f / Z;
        }
    }
    __syncthreads();

    // partial column sums of softmax(attn) for this 16-row strip
    for (int j = tid; j < N_; j += 256) {
        float a = 0.f;
#pragma unroll
        for (int i = 0; i < 16; ++i)
            a += __expf(S[i * 4100 + j] - rowmax[i]) * rowinvZ[i];
        atomicAdd(&colsum[(size_t)b * N_ + j], a);
    }
}

// ---------------- tail kernels ----------------
__global__ void weights_kernel(const float* __restrict__ stk, const float* __restrict__ colsum,
                               float* __restrict__ w, int n) {
    int i = blockIdx.x * blockDim.x + threadIdx.x;
    int stride = gridDim.x * blockDim.x;
    for (; i < n; i += stride) w[i] = stk[i] * colsum[i] * (1.0f / (float)N_);
}
// g[b,c] = sum_j features[b,c,j] * w[b,j]
__global__ __launch_bounds__(256)
void gvec_kernel(const float* __restrict__ features, const float* __restrict__ w,
                 float* __restrict__ g) {
    __shared__ float red[256];
    int bc = blockIdx.x, b = bc >> 9, c = bc & 511;
    const float* fr = features + ((size_t)b * C_ + c) * N_;
    const float* wr = w + (size_t)b * N_;
    float a = 0.f;
    for (int j = threadIdx.x; j < N_; j += 256) a += fr[j] * wr[j];
    red[threadIdx.x] = a; __syncthreads();
    for (int s = 128; s > 0; s >>= 1) {
        if (threadIdx.x < s) red[threadIdx.x] += red[threadIdx.x + s];
        __syncthreads();
    }
    if (threadIdx.x == 0) g[bc] = red[0];
}
// context = Wv @ g ; u = Wf2 @ context
__global__ void ctx_kernel(const float* __restrict__ Wv, const float* __restrict__ g,
                           float* __restrict__ ctx) {
    int i = blockIdx.x * blockDim.x + threadIdx.x;
    if (i >= B_ * C_) return;
    int b = i >> 9, cp = i & 511;
    float a = 0.f;
    for (int c = 0; c < C_; ++c) a += Wv[(size_t)cp * C_ + c] * g[b * C_ + c];
    ctx[i] = a;
}
__global__ void u_kernel(const float* __restrict__ Wf, const float* __restrict__ ctx,
                         float* __restrict__ u) {
    int i = blockIdx.x * blockDim.x + threadIdx.x;
    if (i >= B_ * C_) return;
    int b = i >> 9, o = i & 511;
    float a = 0.f;
    for (int c = 0; c < C_; ++c) a += Wf[(size_t)o * 1024 + 512 + c] * ctx[b * C_ + c];
    u[i] = a;
}
// out[b,o,n] += u[b,o] * rat[b,n]
__global__ void final_kernel(float* __restrict__ out, const float* __restrict__ u,
                             const float* __restrict__ rat) {
    size_t idx = (size_t)blockIdx.x * blockDim.x + threadIdx.x;
    size_t stride = (size_t)gridDim.x * blockDim.x;
    const size_t total = (size_t)B_ * C_ * N_;
    for (; idx < total; idx += stride) {
        int b = (int)(idx >> 21);
        int o = (int)((idx >> 12) & 511);
        int n = (int)(idx & 4095);
        out[idx] += u[b * 512 + o] * rat[b * 4096 + n];
    }
}

extern "C" void kernel_launch(void* const* d_in, const int* in_sizes, int n_in,
                              void* d_out, int out_size, void* d_ws, size_t ws_size,
                              hipStream_t stream) {
    (void)in_sizes; (void)n_in; (void)out_size; (void)ws_size;
    const float* features = (const float*)d_in[0];
    const float* rat      = (const float*)d_in[1];
    const float* Wq       = (const float*)d_in[2];
    const float* Wk       = (const float*)d_in[3];
    const float* Wv       = (const float*)d_in[4];
    const float* Wf       = (const float*)d_in[5];
    float* out = (float*)d_out;

    char* ws = (char*)d_ws;
    size_t off = 0;
    auto alloc = [&](size_t bytes) -> void* {
        void* p = ws + off;
        off += (bytes + 255) & ~(size_t)255;
        return p;
    };
    unsigned short* fbf  = (unsigned short*)alloc((size_t)B_ * C_ * N_ * 2);
    unsigned short* qbf  = (unsigned short*)alloc((size_t)B_ * CQK * N_ * 2);
    unsigned short* kbf  = (unsigned short*)alloc((size_t)B_ * CQK * N_ * 2);
    unsigned short* Wqb  = (unsigned short*)alloc((size_t)CQK * C_ * 2);
    unsigned short* Wkb  = (unsigned short*)alloc((size_t)CQK * C_ * 2);
    unsigned short* Wf1b = (unsigned short*)alloc((size_t)C_ * C_ * 2);
    float* colsum = (float*)alloc((size_t)B_ * N_ * 4);
    float* stk    = (float*)alloc((size_t)B_ * N_ * 4);
    float* wbuf   = (float*)alloc((size_t)B_ * N_ * 4);
    float* g      = (float*)alloc((size_t)B_ * C_ * 4);
    float* ctx    = (float*)alloc((size_t)B_ * C_ * 4);
    float* u      = (float*)alloc((size_t)B_ * C_ * 4);

    cvt_bf16_v8<<<2048, 256, 0, stream>>>(features, fbf, (B_ * C_ * N_) / 8);
    cvt_bf16_v8<<<32, 256, 0, stream>>>(Wq, Wqb, (CQK * C_) / 8);
    cvt_bf16_v8<<<32, 256, 0, stream>>>(Wk, Wkb, (CQK * C_) / 8);
    cvt_wf1_v8<<<128, 256, 0, stream>>>(Wf, Wf1b);
    zero_f32<<<128, 256, 0, stream>>>(colsum, B_ * N_);

    gemm_wmma<true><<<B_ * (CQK / 16) * (N_ / 512), 256, 0, stream>>>(Wqb, fbf, qbf, CQK);
    gemm_wmma<true><<<B_ * (CQK / 16) * (N_ / 512), 256, 0, stream>>>(Wkb, fbf, kbf, CQK);
    gemm_wmma<false><<<B_ * (C_ / 16) * (N_ / 512), 256, 0, stream>>>(Wf1b, fbf, (void*)out, C_);

    size_t smem = (size_t)(128 * 132 + 16 * 132) * 2 + (size_t)16 * 4100 * 4 + 32 * 4;
    attn_kernel<<<B_ * (N_ / 16), 256, smem, stream>>>(qbf, kbf, colsum, stk);

    weights_kernel<<<128, 256, 0, stream>>>(stk, colsum, wbuf, B_ * N_);
    gvec_kernel<<<B_ * C_, 256, 0, stream>>>(features, wbuf, g);
    ctx_kernel<<<16, 256, 0, stream>>>(Wv, g, ctx);
    u_kernel<<<16, 256, 0, stream>>>(Wf, ctx, u);
    final_kernel<<<4096, 256, 0, stream>>>(out, u, rat);
}
